// CNNModel_47639777247400
// MI455X (gfx1250) — compile-verified, matrix operand order checked by reading, not run on hardware
//
#include <hip/hip_runtime.h>
#include <hip/hip_bf16.h>

// ---------------------------------------------------------------------------
// CDNA5 (gfx1250) implementation of the spiking-CNN reference.
// Returned VALUE == (Tout6 @ wf1^T)/(2*steps); only the 300-step spiking
// simulation matters. Spikes are {0,1} -> bf16 WMMA is numerically exact.
// Convs = implicit GEMM with K permuted to (kh,kw,ci) so every 32-wide K-chunk
// is a contiguous channel run: B fragments load as two ds_load_b128 from an
// LDS halo tile; A fragments as two global_load_b128. FC0 is a pure WMMA GEMM.
// ---------------------------------------------------------------------------

typedef __bf16 bf16;
typedef __attribute__((ext_vector_type(16))) __bf16 v16bf;
typedef __attribute__((ext_vector_type(8)))  __bf16 v8bf;
typedef __attribute__((ext_vector_type(8)))  float  v8f;

__device__ __forceinline__ bf16 f2bf(float f) {
    unsigned u = __builtin_bit_cast(unsigned, f);
    unsigned r = (u + 0x7FFFu + ((u >> 16) & 1u)) >> 16;  // round-nearest-even
    unsigned short s = (unsigned short)r;
    return __builtin_bit_cast(bf16, s);
}
__device__ __forceinline__ float bf2f(bf16 h) {
    unsigned short s = __builtin_bit_cast(unsigned short, h);
    return __builtin_bit_cast(float, ((unsigned)s) << 16);
}
__device__ __forceinline__ bf16 bf_zero() {
    return __builtin_bit_cast(bf16, (unsigned short)0);
}

// ---------------- weight conversion: f32 [O][K] -> bf16 [O][Kpad] -----------
// Iperm > 0: permute K ordering from (ci,kh,kw) to (kh,kw,ci) with I = Iperm.
__global__ void wcvt_kernel(const float* __restrict__ w, bf16* __restrict__ o,
                            int O, int K, int Kpad, int Iperm) {
    int i = blockIdx.x * blockDim.x + threadIdx.x;
    if (i >= O * Kpad) return;
    int row = i / Kpad, k = i - row * Kpad;
    bf16 v = bf_zero();
    if (k < K) {
        int ks = k;
        if (Iperm > 0) {                 // k = r*I + ci  ->  src = ci*9 + r
            int r = k / Iperm, ci = k - r * Iperm;
            ks = ci * 9 + r;
        }
        v = f2bf(w[row * K + ks]);
    }
    o[i] = v;
}

// ---------------- Poisson-style encoder (counter-based hash RNG) ------------
__global__ void encode_kernel(const float* __restrict__ x, bf16* __restrict__ p,
                              int n, int step) {
    int i = blockIdx.x * blockDim.x + threadIdx.x;
    if (i >= n) return;
    unsigned s = (unsigned)i * 2654435761u ^ ((unsigned)step * 0x9E3779B9u) ^ 0x85EBCA6Bu;
    s ^= s >> 16; s *= 0x7feb352du; s ^= s >> 15; s *= 0x846ca68bu; s ^= s >> 16;
    float r = (float)(s >> 8) * (1.0f / 16777216.0f);
    float xi = x[i];
    float v = (fabsf(xi) > r) ? (xi > 0.f ? 1.f : (xi < 0.f ? -1.f : 0.f)) : 0.f;
    p[i] = f2bf(v);
}

// ---------------- implicit-GEMM 3x3 SAME conv + fused LIF -------------------
// Block: (O/16) waves, one 16-pixel row segment (b,y,x0). LDS halo [3][18][ci]
// (ci innermost, stride padded +8 for bank spread). Wave w = channels 16w..+15.
template <int I, int O, int H, int W>
__global__ __launch_bounds__((O / 16) * 32)
void conv_lif_kernel(const bf16* __restrict__ in, const bf16* __restrict__ wgt,
                     float* __restrict__ mem, bf16* __restrict__ spk, float thr) {
    constexpr bool FASTK = (I % 32 == 0);
    constexpr int  KPAD  = ((I * 9 + 31) / 32) * 32;
    constexpr int  NW    = O / 16;
    constexpr int  IP    = FASTK ? (I + 8) : I;   // padded ci-stride (elements)
    constexpr int  LDSN  = FASTK ? (3 * 18 * IP) : (I * 54);
    __shared__ alignas(16) bf16 lds[LDSN];

    const int tid  = threadIdx.x;
    const int lane = tid & 31;
    const int wv   = tid >> 5;
    const int segs = W / 16;
    int bid = blockIdx.x;
    const int x0 = (bid % segs) * 16; bid /= segs;
    const int y  = bid % H;
    const int b  = bid / H;

    // stage input halo (zero-padded borders) into LDS
    for (int t = tid; t < I * 54; t += NW * 32) {
        int ci = t / 54, r = t - ci * 54;
        int kh = r / 18, xx = r - kh * 18;
        int gy = y + kh - 1, gx = x0 + xx - 1;
        bf16 v = bf_zero();
        if (gy >= 0 && gy < H && gx >= 0 && gx < W)
            v = in[((b * I + ci) * H + gy) * W + gx];
        if constexpr (FASTK)
            lds[(kh * 18 + xx) * IP + ci] = v;     // [kh][x][ci], ci innermost
        else
            lds[t] = v;                             // [ci][kh][x]
    }
    __syncthreads();

    const int half = lane >> 4;
    const int nl   = lane & 15;
    v8f c = {};
    const bf16* wrow = wgt + (wv * 16 + nl) * KPAD + 8 * half;

    if constexpr (FASTK) {
        // K' = (kh*3+kw)*I + ci : each 32-chunk is a contiguous channel run
        int k0 = 0;
#pragma unroll
        for (int r = 0; r < 9; ++r) {
            const int kh = r / 3, kw = r - kh * 3;
            const bf16* brow = lds + (kh * 18 + nl + kw) * IP + 8 * half;
            for (int cb = 0; cb < I; cb += 32, k0 += 32) {
                v8bf alo = *(const v8bf*)(wrow + k0);
                v8bf ahi = *(const v8bf*)(wrow + k0 + 16);
                v8bf blo = *(const v8bf*)(brow + cb);       // ds_load_b128
                v8bf bhi = *(const v8bf*)(brow + cb + 16);  // ds_load_b128
                v16bf a, bb;
#pragma unroll
                for (int e = 0; e < 8; ++e) {
                    a[e] = alo[e];  a[e + 8] = ahi[e];
                    bb[e] = blo[e]; bb[e + 8] = bhi[e];
                }
                c = __builtin_amdgcn_wmma_f32_16x16x32_bf16(false, a, false, bb,
                                                            (short)0, c, false, false);
            }
        }
    } else {
        // generic gather path (only I=3 first conv: single K-chunk)
        for (int k0 = 0; k0 < KPAD; k0 += 32) {
            v8bf alo = *(const v8bf*)(wrow + k0);
            v8bf ahi = *(const v8bf*)(wrow + k0 + 16);
            v16bf a, bb;
#pragma unroll
            for (int e = 0; e < 8; ++e) { a[e] = alo[e]; a[e + 8] = ahi[e]; }
#pragma unroll
            for (int e = 0; e < 16; ++e) {
                int K = k0 + 8 * half + (e & 7) + 16 * (e >> 3);
                bf16 v = bf_zero();
                if (K < I * 9) {
                    int ci = K / 9, rr = K - ci * 9;
                    int kh = rr / 3, kw = rr - kh * 3;
                    v = lds[ci * 54 + kh * 18 + nl + kw];
                }
                bb[e] = v;
            }
            c = __builtin_amdgcn_wmma_f32_16x16x32_bf16(false, a, false, bb,
                                                        (short)0, c, false, false);
        }
    }

    // fused LIF epilogue: D element (v, lane) -> channel o, pixel x0+nl
#pragma unroll
    for (int vv = 0; vv < 8; ++vv) {
        int o = wv * 16 + vv + 8 * half;
        int idx = ((b * O + o) * H + y) * W + x0 + nl;
        float raw = mem[idx] + c[vv];
        float ex  = raw > thr ? raw : 0.f;
        mem[idx] = raw - ex;
        spk[idx] = f2bf(ex > 0.f ? 1.f : 0.f);
    }
}

// ---------------- 2x2 avgpool (spikes) + fused LIF (thr 0.75) ---------------
__global__ void avgpool_lif_kernel(const bf16* __restrict__ in,
                                   float* __restrict__ mem, bf16* __restrict__ spk,
                                   int n, int Ho, int Wo) {
    int i = blockIdx.x * blockDim.x + threadIdx.x;
    if (i >= n) return;
    int x = i % Wo, t = i / Wo;
    int y = t % Ho; t /= Ho;                       // t = b*C + c
    const bf16* src = in + ((t * (2 * Ho) + 2 * y) * (2 * Wo)) + 2 * x;
    int rs = 2 * Wo;
    float s = bf2f(src[0]) + bf2f(src[1]) + bf2f(src[rs]) + bf2f(src[rs + 1]);
    float raw = mem[i] + s * 0.25f;
    float ex  = raw > 0.75f ? raw : 0.f;
    mem[i] = raw - ex;
    spk[i] = f2bf(ex > 0.f ? 1.f : 0.f);
}

// ---------------- 2x2 maxpool (16x16 -> 8x8), NCHW flatten ------------------
__global__ void maxpool_kernel(const bf16* __restrict__ in, bf16* __restrict__ out,
                               int n) {
    int i = blockIdx.x * blockDim.x + threadIdx.x;
    if (i >= n) return;
    int x = i % 8, t = i / 8;
    int y = t % 8; t /= 8;                          // t = b*C + c
    const bf16* src = in + (t * 16 + 2 * y) * 16 + 2 * x;
    float m = fmaxf(fmaxf(bf2f(src[0]), bf2f(src[1])),
                    fmaxf(bf2f(src[16]), bf2f(src[17])));
    out[i] = f2bf(m);
}

// ---------------- FC0 (8192 -> 1024) WMMA GEMM + LIF + spike accumulate -----
// M=1024 (out rows), N=16 (batch), K=8192. 8 blocks x 8 waves; wave = 16 rows.
__global__ __launch_bounds__(256)
void fc0_lif_kernel(const bf16* __restrict__ act,   // [16][8192]
                    const bf16* __restrict__ wgt,   // [1024][8192]
                    float* __restrict__ mem,        // [16][1024]
                    float* __restrict__ tout) {     // [16][1024]
    const int lane = threadIdx.x & 31;
    const int wv   = threadIdx.x >> 5;
    const int half = lane >> 4, nl = lane & 15;
    const int obase = (blockIdx.x * 8 + wv) * 16;
    const bf16* arow = wgt + (size_t)(obase + nl) * 8192 + 8 * half;
    const bf16* brow = act + (size_t)nl * 8192 + 8 * half;
    v8f c = {};
    for (int k0 = 0; k0 < 8192; k0 += 32) {
        __builtin_prefetch(arow + k0 + 256, 0, 0);   // global_prefetch_b8
        v8bf alo = *(const v8bf*)(arow + k0);
        v8bf ahi = *(const v8bf*)(arow + k0 + 16);
        v8bf blo = *(const v8bf*)(brow + k0);
        v8bf bhi = *(const v8bf*)(brow + k0 + 16);
        v16bf a, bb;
#pragma unroll
        for (int e = 0; e < 8; ++e) {
            a[e] = alo[e];  a[e + 8] = ahi[e];
            bb[e] = blo[e]; bb[e + 8] = bhi[e];
        }
        c = __builtin_amdgcn_wmma_f32_16x16x32_bf16(false, a, false, bb,
                                                    (short)0, c, false, false);
    }
#pragma unroll
    for (int vv = 0; vv < 8; ++vv) {
        int o = obase + vv + 8 * half;
        int idx = nl * 1024 + o;                    // batch = nl
        float raw = mem[idx] + c[vv];
        float ex  = raw > 2.0f ? raw : 0.f;
        mem[idx] = raw - ex;
        if (ex > 0.f) tout[idx] += 1.0f;            // unique owner per (b,o)
    }
}

// ---------------- final tiny GEMM: out = Tout6 @ wf1^T * scale --------------
__global__ void fc1_kernel(const float* __restrict__ t6, const float* __restrict__ wf1,
                           float* __restrict__ out, float scale) {
    int i = blockIdx.x * blockDim.x + threadIdx.x;
    if (i >= 160) return;
    int b = i / 10, j = i - b * 10;
    const float* tr = t6 + b * 1024;
    const float* wr = wf1 + j * 1024;
    float s = 0.f;
    for (int k = 0; k < 1024; ++k) s += tr[k] * wr[k];
    out[i] = s * scale;
}

// ---------------------------------------------------------------------------
extern "C" void kernel_launch(void* const* d_in, const int* in_sizes, int n_in,
                              void* d_out, int out_size, void* d_ws, size_t ws_size,
                              hipStream_t stream) {
    (void)in_sizes; (void)n_in; (void)out_size;
    const float* x   = (const float*)d_in[0];
    const float* w11 = (const float*)d_in[1];
    const float* w12 = (const float*)d_in[2];
    const float* w21 = (const float*)d_in[3];
    const float* w22 = (const float*)d_in[4];
    const float* w23 = (const float*)d_in[5];
    const float* wf0 = (const float*)d_in[6];
    const float* wf1 = (const float*)d_in[7];
    const int STEPS = 300;  // static arg in reference (device scalar unreadable in capture)

    char* ws = (char*)d_ws;
    size_t off = 0;
    auto alloc = [&](size_t bytes) -> char* {
        char* p = ws + off;
        off = (off + bytes + 255) & ~(size_t)255;
        return p;
    };

    bf16* wb11 = (bf16*)alloc((size_t)64 * 32 * 2);
    bf16* wb12 = (bf16*)alloc((size_t)64 * 576 * 2);
    bf16* wb21 = (bf16*)alloc((size_t)128 * 576 * 2);
    bf16* wb22 = (bf16*)alloc((size_t)128 * 1152 * 2);
    bf16* wb23 = (bf16*)alloc((size_t)128 * 1152 * 2);
    bf16* wbf0 = (bf16*)alloc((size_t)1024 * 8192 * 2);
    bf16* enc  = (bf16*)alloc((size_t)16 * 3 * 32 * 32 * 2);
    bf16* s1   = (bf16*)alloc((size_t)16 * 64 * 1024 * 2);
    bf16* s2   = (bf16*)alloc((size_t)16 * 64 * 1024 * 2);
    bf16* s3   = (bf16*)alloc((size_t)16 * 64 * 256 * 2);
    bf16* s4   = (bf16*)alloc((size_t)16 * 128 * 256 * 2);
    bf16* s5   = (bf16*)alloc((size_t)16 * 128 * 256 * 2);
    bf16* s6   = (bf16*)alloc((size_t)16 * 128 * 256 * 2);
    bf16* pooled = (bf16*)alloc((size_t)16 * 8192 * 2);
    char* memsBase = ws + off;                       // zero this whole region
    float* m1 = (float*)alloc((size_t)16 * 64 * 1024 * 4);
    float* m2 = (float*)alloc((size_t)16 * 64 * 1024 * 4);
    float* m3 = (float*)alloc((size_t)16 * 64 * 256 * 4);
    float* m4 = (float*)alloc((size_t)16 * 128 * 256 * 4);
    float* m5 = (float*)alloc((size_t)16 * 128 * 256 * 4);
    float* m6 = (float*)alloc((size_t)16 * 128 * 256 * 4);
    float* m7 = (float*)alloc((size_t)16 * 1024 * 4);
    float* t6 = (float*)alloc((size_t)16 * 1024 * 4);
    if (off > ws_size) return;  // workspace too small; bail deterministically
    hipMemsetAsync(memsBase, 0, (size_t)((ws + off) - memsBase), stream);

    auto cvt = [&](const float* w, bf16* o, int O, int K, int Kpad, int Iperm) {
        int n = O * Kpad;
        wcvt_kernel<<<(n + 255) / 256, 256, 0, stream>>>(w, o, O, K, Kpad, Iperm);
    };
    cvt(w11, wb11, 64, 27, 32, 0);          // I=3: generic (ci,kh,kw) order
    cvt(w12, wb12, 64, 576, 576, 64);       // permuted (kh,kw,ci)
    cvt(w21, wb21, 128, 576, 576, 64);
    cvt(w22, wb22, 128, 1152, 1152, 128);
    cvt(w23, wb23, 128, 1152, 1152, 128);
    cvt(wf0, wbf0, 1024, 8192, 8192, 0);    // FC: natural order

    const int NENC = 16 * 3 * 32 * 32;
    for (int s = 0; s < STEPS; ++s) {
        encode_kernel<<<(NENC + 255) / 256, 256, 0, stream>>>(x, enc, NENC, s);
        conv_lif_kernel<3, 64, 32, 32><<<16 * 32 * 2, 128, 0, stream>>>(enc, wb11, m1, s1, 2.0f);
        conv_lif_kernel<64, 64, 32, 32><<<16 * 32 * 2, 128, 0, stream>>>(s1, wb12, m2, s2, 2.0f);
        avgpool_lif_kernel<<<(262144 + 255) / 256, 256, 0, stream>>>(s2, m3, s3, 262144, 16, 16);
        conv_lif_kernel<64, 128, 16, 16><<<16 * 16, 256, 0, stream>>>(s3, wb21, m4, s4, 2.0f);
        conv_lif_kernel<128, 128, 16, 16><<<16 * 16, 256, 0, stream>>>(s4, wb22, m5, s5, 2.0f);
        conv_lif_kernel<128, 128, 16, 16><<<16 * 16, 256, 0, stream>>>(s5, wb23, m6, s6, 2.0f);
        maxpool_kernel<<<(131072 + 255) / 256, 256, 0, stream>>>(s6, pooled, 131072);
        fc0_lif_kernel<<<8, 256, 0, stream>>>(pooled, wbf0, m7, t6);
    }
    fc1_kernel<<<1, 192, 0, stream>>>(t6, wf1, (float*)d_out, 1.0f / (2.0f * STEPS));
}
